// QKV_Write_65189013618891
// MI455X (gfx1250) — matrix operand order repaired and checked by reading, not compiled, not run on hardware
//
#include <hip/hip_runtime.h>
#include <hip/hip_bf16.h>

// ---------------------------------------------------------------------------
// Perceiver cross-attention block on MI455X (gfx1250, wave32, WMMA bf16)
//   B=256, DIM=1024, H=16, HD=64, NL=64, NKV=256, DS=4
// ---------------------------------------------------------------------------

typedef __bf16 bf16;
typedef __attribute__((ext_vector_type(16))) __bf16 v16bf;
typedef __attribute__((ext_vector_type(8)))  __bf16 v8bf;
typedef __attribute__((ext_vector_type(8)))  float  v8f;
typedef __attribute__((ext_vector_type(4)))  int    v4i;

#define BATCH 256
#define DIMC  1024
#define HEADS 16
#define HDIM  64
#define NLQ   64
#define NKVT  256
#define DSC   4
#define LNEPS 1e-5f

// ---- CDNA5 async global->LDS path (GLOBAL_LOAD_ASYNC_TO_LDS_B128) --------
#if defined(__has_builtin)
#if __has_builtin(__builtin_amdgcn_global_load_async_to_lds_b128)
#define ASYNC_G2S 1
#endif
#endif

#if defined(ASYNC_G2S)
typedef __attribute__((address_space(1))) v4i as1_v4i;  // global int4*
typedef __attribute__((address_space(3))) v4i as3_v4i;  // LDS int4*
__device__ __forceinline__ as1_v4i* as_global(const void* p) {
  // flat global address == AS1 address (value-preserving)
  return (as1_v4i*)(unsigned long long)p;
}
__device__ __forceinline__ as3_v4i* as_lds(void* p) {
  // LDS aperture maps flat[31:0] -> LDS byte address (ISA 10.2)
  return (as3_v4i*)(unsigned int)(unsigned long long)p;
}
__device__ __forceinline__ void wait_async0() {
#if __has_builtin(__builtin_amdgcn_s_wait_asynccnt)
  __builtin_amdgcn_s_wait_asynccnt(0);
#else
  asm volatile("s_wait_asynccnt 0x0" ::: "memory");
#endif
}
#endif

__device__ __forceinline__ v16bf mk16(v8bf lo, v8bf hi) {
  return __builtin_shufflevector(lo, hi, 0, 1, 2, 3, 4, 5, 6, 7,
                                         8, 9, 10, 11, 12, 13, 14, 15);
}

__device__ __forceinline__ v8f wmma_bf16(v16bf a, v16bf b, v8f c) {
  // D = A(16x32 bf16) x B(32x16 bf16) + C(16x16 f32) -> v_wmma_f32_16x16x32_bf16
  return __builtin_amdgcn_wmma_f32_16x16x32_bf16(false, a, false, b,
                                                 (short)0, c, false, false);
}

__device__ __forceinline__ float wave_sum(float v) {
#pragma unroll
  for (int off = 16; off > 0; off >>= 1) v += __shfl_xor(v, off, 32);
  return v;
}

// ---------------------------------------------------------------------------
// Kernel 1: fused LayerNorm (per 1024-wide row) + cast to bf16.
// ---------------------------------------------------------------------------
__global__ __launch_bounds__(256) void ln_bf16_kernel(
    const float* __restrict__ src, const float* __restrict__ gamma,
    const float* __restrict__ beta, bf16* __restrict__ dst) {
  const int row  = blockIdx.x;
  const int tid  = threadIdx.x;
  const int lane = tid & 31;
  const int wid  = tid >> 5;
  const float* p = src + (size_t)row * DIMC;

  float v[4];
  float s = 0.f;
#pragma unroll
  for (int i = 0; i < 4; ++i) { v[i] = p[tid + i * 256]; s += v[i]; }

  __shared__ float red[8];
  s = wave_sum(s);
  if (lane == 0) red[wid] = s;
  __syncthreads();
  float tot = 0.f;
#pragma unroll
  for (int w = 0; w < 8; ++w) tot += red[w];
  const float mean = tot * (1.f / DIMC);
  __syncthreads();  // before reusing red[]

  float vs = 0.f;
#pragma unroll
  for (int i = 0; i < 4; ++i) { float d = v[i] - mean; vs += d * d; }
  vs = wave_sum(vs);
  if (lane == 0) red[wid] = vs;
  __syncthreads();
  tot = 0.f;
#pragma unroll
  for (int w = 0; w < 8; ++w) tot += red[w];
  const float rstd = rsqrtf(tot * (1.f / DIMC) + LNEPS);

  bf16* out = dst + (size_t)row * DIMC;
#pragma unroll
  for (int i = 0; i < 4; ++i) {
    int c = tid + i * 256;
    out[c] = (bf16)((v[i] - mean) * rstd * gamma[c] + beta[c]);
  }
}

// ---------------------------------------------------------------------------
// Kernel 2: fp32 -> bf16 cast (weights). grid*1024 == n.
// ---------------------------------------------------------------------------
__global__ __launch_bounds__(256) void cast_bf16_kernel(
    const float* __restrict__ s, bf16* __restrict__ d) {
  int i = blockIdx.x * 1024 + threadIdx.x;
#pragma unroll
  for (int k = 0; k < 4; ++k) d[i + k * 256] = (bf16)s[i + k * 256];
}

// ---------------------------------------------------------------------------
// Kernel 3: C[M x 1024] = A[M x 1024] @ W[1024 x 1024]^T   (bf16 in)
// 128x128 block tile, K-step 32, 8 waves (4M x 2N), 32x64 wave tile.
// Double-buffered LDS fed by GLOBAL_LOAD_ASYNC_TO_LDS_B128 when available
// (ASYNCcnt + barrier), otherwise register-staged global->ds pipeline.
// ---------------------------------------------------------------------------
template <bool F32OUT>
__global__ __launch_bounds__(256) void gemm_wmma_kernel(
    const bf16* __restrict__ A, const bf16* __restrict__ W,
    bf16* __restrict__ Cb, float* __restrict__ Cf,
    const float* __restrict__ bias, const float* __restrict__ resid) {
  const int tid  = threadIdx.x;
  const int lane = tid & 31;
  const int wid  = tid >> 5;
  const int wm   = wid >> 1;          // 0..3
  const int wn   = wid & 1;           // 0..1
  const int m0   = blockIdx.y * 128;
  const int n0   = blockIdx.x * 128;

  __shared__ bf16 As[2][128][40];     // +8 pad, rows stay 16B-aligned
  __shared__ bf16 Bs[2][128][40];

  const int r    = lane & 15;
  const int hi8  = (lane & 16) ? 8 : 0;
  const int hi16 = (lane & 16) ? 16 : 0;

  const int c0  = tid,          row0 = c0 >> 2, seg0 = (c0 & 3) * 8;
  const int c1  = tid + 256,    row1 = c1 >> 2, seg1 = (c1 & 3) * 8;

  v8f acc[2][4];
#pragma unroll
  for (int mt = 0; mt < 2; ++mt)
#pragma unroll
    for (int nt = 0; nt < 4; ++nt) acc[mt][nt] = {};

#if defined(ASYNC_G2S)
  // -------- async DMA-style global->LDS (no VGPR staging) -----------------
  auto g2s_async = [&](int buf, int k0) {
    __builtin_amdgcn_global_load_async_to_lds_b128(
        as_global(A + (size_t)(m0 + row0) * DIMC + k0 + seg0),
        as_lds(&As[buf][row0][seg0]), 0, 0);
    __builtin_amdgcn_global_load_async_to_lds_b128(
        as_global(W + (size_t)(n0 + row0) * DIMC + k0 + seg0),
        as_lds(&Bs[buf][row0][seg0]), 0, 0);
    __builtin_amdgcn_global_load_async_to_lds_b128(
        as_global(A + (size_t)(m0 + row1) * DIMC + k0 + seg1),
        as_lds(&As[buf][row1][seg1]), 0, 0);
    __builtin_amdgcn_global_load_async_to_lds_b128(
        as_global(W + (size_t)(n0 + row1) * DIMC + k0 + seg1),
        as_lds(&Bs[buf][row1][seg1]), 0, 0);
  };
  g2s_async(0, 0);
  wait_async0();
  __syncthreads();
#else
  // -------- fallback: register-staged pipeline ----------------------------
#pragma unroll
  for (int i = 0; i < 2; ++i) {
    int c = tid + i * 256, row = c >> 2, seg = (c & 3) * 8;
    *(v8bf*)&As[0][row][seg] = *(const v8bf*)(A + (size_t)(m0 + row) * DIMC + seg);
    *(v8bf*)&Bs[0][row][seg] = *(const v8bf*)(W + (size_t)(n0 + row) * DIMC + seg);
  }
  __syncthreads();
#endif

  for (int ks = 0; ks < 32; ++ks) {
    const int cur = ks & 1;
    const bool havenext = (ks + 1) < 32;

#if defined(ASYNC_G2S)
    if (havenext) {
      g2s_async(cur ^ 1, (ks + 1) * 32);   // overlap with WMMA below
      if ((ks + 2) < 32) {                 // warm L2 two tiles ahead
        __builtin_prefetch(A + (size_t)(m0 + row0) * DIMC + (ks + 2) * 32 + seg0, 0, 1);
        __builtin_prefetch(W + (size_t)(n0 + row0) * DIMC + (ks + 2) * 32 + seg0, 0, 1);
      }
    }
#else
    v8bf ra[2], rb[2];
    if (havenext) {
      const int k0 = (ks + 1) * 32;
#pragma unroll
      for (int i = 0; i < 2; ++i) {
        int c = tid + i * 256, row = c >> 2, seg = (c & 3) * 8;
        const bf16* ap = A + (size_t)(m0 + row) * DIMC + k0 + seg;
        const bf16* bp = W + (size_t)(n0 + row) * DIMC + k0 + seg;
        ra[i] = *(const v8bf*)ap;
        rb[i] = *(const v8bf*)bp;
        if (i == 0 && (ks + 2) < 32) {
          __builtin_prefetch(ap + 32, 0, 1);
          __builtin_prefetch(bp + 32, 0, 1);
        }
      }
    }
#endif

    v16bf af[2], bfb[4];
#pragma unroll
    for (int mt = 0; mt < 2; ++mt) {
      const bf16* p = &As[cur][wm * 32 + mt * 16 + r][0];
      af[mt] = mk16(*(const v8bf*)(p + hi8), *(const v8bf*)(p + 16 + hi8));
    }
#pragma unroll
    for (int nt = 0; nt < 4; ++nt) {
      const bf16* p = &Bs[cur][wn * 64 + nt * 16 + r][0];
      bfb[nt] = mk16(*(const v8bf*)(p + hi8), *(const v8bf*)(p + 16 + hi8));
    }
#pragma unroll
    for (int mt = 0; mt < 2; ++mt)
#pragma unroll
      for (int nt = 0; nt < 4; ++nt)
        acc[mt][nt] = wmma_bf16(af[mt], bfb[nt], acc[mt][nt]);

    if (havenext) {
#if defined(ASYNC_G2S)
      wait_async0();
      __syncthreads();
#else
      const int nxt = cur ^ 1;
#pragma unroll
      for (int i = 0; i < 2; ++i) {
        int c = tid + i * 256, row = c >> 2, seg = (c & 3) * 8;
        *(v8bf*)&As[nxt][row][seg] = ra[i];
        *(v8bf*)&Bs[nxt][row][seg] = rb[i];
      }
      __syncthreads();
#endif
    }
  }

  // epilogue: D layout = lane(l&15)=col, vgpr vr -> row vr (+8 for lanes>=16)
#pragma unroll
  for (int mt = 0; mt < 2; ++mt)
#pragma unroll
    for (int nt = 0; nt < 4; ++nt)
#pragma unroll
      for (int vr = 0; vr < 8; ++vr) {
        const int m = m0 + wm * 32 + mt * 16 + vr + hi8;
        const int n = n0 + wn * 64 + nt * 16 + r;
        if constexpr (F32OUT) {
          Cf[(size_t)m * DIMC + n] =
              acc[mt][nt][vr] + bias[n] + resid[(size_t)m * DIMC + n];
        } else {
          Cb[(size_t)m * DIMC + n] = (bf16)acc[mt][nt][vr];
        }
      }
}

// ---------------------------------------------------------------------------
// Kernel 4: per-(b,h) attention. S = QK^T/8 (WMMA), block-causal mask
// (col < (row+1)*DS), stable softmax in LDS, O = P@V (WMMA, V staged
// transposed in LDS). All tile-skip bounds are forced wave-uniform through
// readfirstlane so WMMA never sits under EXEC-masked control flow; K-tile
// loads are software-pipelined against the WMMAs.
// ---------------------------------------------------------------------------
__global__ __launch_bounds__(256) void attn_wmma_kernel(
    const bf16* __restrict__ Q, const bf16* __restrict__ Kp,
    const bf16* __restrict__ Vp, float* __restrict__ O) {
  const int h = blockIdx.x, b = blockIdx.y;
  const int tid  = threadIdx.x;
  const int lane = tid & 31;
  const int r    = lane & 15;
  const int hi8  = (lane & 16) ? 8 : 0;
  const int hi16 = (lane & 16) ? 16 : 0;

  // wave-uniform scalars (SALU control flow; EXEC stays all-ones)
  const int wid   = __builtin_amdgcn_readfirstlane(tid >> 5);
  const int mtile = wid >> 1;          // 4 M-tiles of 16 query rows
  const int nhalf = wid & 1;

  __shared__ float Ssm[64][260];      // scores (f32)
  __shared__ bf16  Pm[64][264];       // softmax probs (bf16)
  __shared__ bf16  Vt[64][264];       // V transposed: Vt[d][j]
  __shared__ float rred[64][4];

  // ---- stage V^T: thread t copies kv-row j=t (64 bf16) into column t ----
  {
    const bf16* vrow = Vp + ((size_t)(b * NKVT + tid)) * DIMC + h * HDIM;
#pragma unroll
    for (int dc = 0; dc < 8; ++dc) {
      v8bf vv = *(const v8bf*)(vrow + dc * 8);
#pragma unroll
      for (int e = 0; e < 8; ++e) Vt[dc * 8 + e][tid] = vv[e];
    }
  }

  // ---- S = Q K^T * HD^-0.5 ----
  v16bf qf[2];
#pragma unroll
  for (int ksq = 0; ksq < 2; ++ksq) {
    const bf16* qrow =
        Q + ((size_t)(b * NLQ + mtile * 16 + r)) * DIMC + h * HDIM + ksq * 32;
    qf[ksq] = mk16(*(const v8bf*)(qrow + hi8), *(const v8bf*)(qrow + 16 + hi8));
  }
  const int ntlim = (mtile + 1) * 4;   // tiles >= this are fully masked
  const int nt0   = nhalf * 8;
  const int ntend = (ntlim < nt0 + 8) ? ntlim : (nt0 + 8);  // uniform

  auto load_kf = [&](int ntile, v16bf kf[2]) {
#pragma unroll
    for (int ksq = 0; ksq < 2; ++ksq) {
      const bf16* krow = Kp + ((size_t)(b * NKVT + ntile * 16 + r)) * DIMC +
                         h * HDIM + ksq * 32;
      kf[ksq] = mk16(*(const v8bf*)(krow + hi8), *(const v8bf*)(krow + 16 + hi8));
    }
  };

  v16bf kf[2];
  if (nt0 < ntend) load_kf(nt0, kf);
  for (int ntile = nt0; ntile < ntend; ++ntile) {
    const v16bf kc0 = kf[0], kc1 = kf[1];
    if (ntile + 1 < ntend) load_kf(ntile + 1, kf);  // overlap next loads
    v8f acc = {};
    acc = wmma_bf16(qf[0], kc0, acc);
    acc = wmma_bf16(qf[1], kc1, acc);
#pragma unroll
    for (int vr = 0; vr < 8; ++vr)
      Ssm[mtile * 16 + vr + hi8][ntile * 16 + r] = acc[vr] * 0.125f;
  }
  __syncthreads();

  // ---- masked, numerically-stable softmax (4 threads per row) ----
  {
    const int row = tid >> 2, seg = tid & 3;
    const int cb = seg * 64;
    const int limit = (row + 1) * DSC;
    float mx = -3.4e38f;
    for (int c = cb; c < cb + 64; ++c)
      if (c < limit) mx = fmaxf(mx, Ssm[row][c]);
    rred[row][seg] = mx;
    __syncthreads();
    mx = fmaxf(fmaxf(rred[row][0], rred[row][1]),
               fmaxf(rred[row][2], rred[row][3]));
    __syncthreads();
    float s = 0.f;
    for (int c = cb; c < cb + 64; ++c)
      if (c < limit) s += __expf(Ssm[row][c] - mx);
    rred[row][seg] = s;
    __syncthreads();
    s = rred[row][0] + rred[row][1] + rred[row][2] + rred[row][3];
    const float inv = 1.f / s;
    for (int c = cb; c < cb + 64; ++c)
      Pm[row][c] = (c < limit) ? (bf16)(__expf(Ssm[row][c] - mx) * inv)
                               : (bf16)0.0f;
  }
  __syncthreads();

  // ---- O = P @ V ----
  const int klim = (mtile + 1) * 2;    // P rows are zero past (mtile+1)*64
#pragma unroll
  for (int ntp = 0; ntp < 2; ++ntp) {
    const int ntile = nhalf * 2 + ntp;  // 4 d-tiles of 16
    v8f acc = {};
    for (int ks = 0; ks < klim; ++ks) {  // uniform bound
      const bf16* prow = &Pm[mtile * 16 + r][ks * 32];
      v16bf pf = mk16(*(const v8bf*)(prow + hi8), *(const v8bf*)(prow + 16 + hi8));
      const bf16* vtr = &Vt[ntile * 16 + r][ks * 32];
      v16bf vf = mk16(*(const v8bf*)(vtr + hi8), *(const v8bf*)(vtr + 16 + hi8));
      acc = wmma_bf16(pf, vf, acc);
    }
#pragma unroll
    for (int vr = 0; vr < 8; ++vr)
      O[((size_t)(b * NLQ + mtile * 16 + vr + hi8)) * DIMC + h * HDIM +
        ntile * 16 + r] = acc[vr];
  }
}

// ---------------------------------------------------------------------------
// host-side orchestration
// ---------------------------------------------------------------------------
extern "C" void kernel_launch(void* const* d_in, const int* in_sizes, int n_in,
                              void* d_out, int out_size, void* d_ws,
                              size_t ws_size, hipStream_t stream) {
  (void)in_sizes; (void)n_in; (void)out_size; (void)ws_size;
  const float* x     = (const float*)d_in[0];
  const float* lat   = (const float*)d_in[1];
  const float* Wq    = (const float*)d_in[2];
  const float* Wk    = (const float*)d_in[3];
  const float* Wv    = (const float*)d_in[4];
  const float* Wo    = (const float*)d_in[5];
  const float* bo    = (const float*)d_in[6];
  const float* g_out = (const float*)d_in[7];
  const float* b_out = (const float*)d_in[8];
  const float* g_x   = (const float*)d_in[9];
  const float* b_x   = (const float*)d_in[10];
  const float* g_l   = (const float*)d_in[11];
  const float* b_l   = (const float*)d_in[12];
  float* out = (float*)d_out;

  const int RQ  = BATCH * NLQ;    // 16384 query rows
  const int RKV = BATCH * NKVT;   // 65536 kv rows

  char* ws = (char*)d_ws;
  size_t off = 0;
  auto take = [&](size_t bytes) -> char* {
    char* p = ws + off;
    off = (off + bytes + 255) & ~(size_t)255;
    return p;
  };
  bf16* lnx  = (bf16*)take((size_t)RKV * DIMC * sizeof(bf16));
  bf16* lnl  = (bf16*)take((size_t)RQ  * DIMC * sizeof(bf16));
  bf16* wqb  = (bf16*)take((size_t)DIMC * DIMC * sizeof(bf16));
  bf16* wkb  = (bf16*)take((size_t)DIMC * DIMC * sizeof(bf16));
  bf16* wvb  = (bf16*)take((size_t)DIMC * DIMC * sizeof(bf16));
  bf16* wob  = (bf16*)take((size_t)DIMC * DIMC * sizeof(bf16));
  bf16* Qb   = (bf16*)take((size_t)RQ  * DIMC * sizeof(bf16));
  bf16* Kb   = (bf16*)take((size_t)RKV * DIMC * sizeof(bf16));
  bf16* Vb   = (bf16*)take((size_t)RKV * DIMC * sizeof(bf16));
  float* Ao  = (float*)take((size_t)RQ * DIMC * sizeof(float));
  bf16* Aol  = (bf16*)take((size_t)RQ * DIMC * sizeof(bf16));

  // 1) pre-norms (fused fp32->bf16)
  ln_bf16_kernel<<<RKV, 256, 0, stream>>>(x, g_x, b_x, lnx);
  ln_bf16_kernel<<<RQ, 256, 0, stream>>>(lat, g_l, b_l, lnl);

  // 2) weight casts
  cast_bf16_kernel<<<1024, 256, 0, stream>>>(Wq, wqb);
  cast_bf16_kernel<<<1024, 256, 0, stream>>>(Wk, wkb);
  cast_bf16_kernel<<<1024, 256, 0, stream>>>(Wv, wvb);
  cast_bf16_kernel<<<1024, 256, 0, stream>>>(Wo, wob);

  // 3) projections: X @ W^T (WMMA bf16)
  gemm_wmma_kernel<false><<<dim3(DIMC / 128, RQ / 128), 256, 0, stream>>>(
      lnl, wqb, Qb, nullptr, nullptr, nullptr);
  gemm_wmma_kernel<false><<<dim3(DIMC / 128, RKV / 128), 256, 0, stream>>>(
      lnx, wkb, Kb, nullptr, nullptr, nullptr);
  gemm_wmma_kernel<false><<<dim3(DIMC / 128, RKV / 128), 256, 0, stream>>>(
      lnx, wvb, Vb, nullptr, nullptr, nullptr);

  // 4) attention (one block per (b,h))
  attn_wmma_kernel<<<dim3(HEADS, BATCH), 256, 0, stream>>>(Qb, Kb, Vb, Ao);

  // 5) norm_out (fused fp32->bf16)
  ln_bf16_kernel<<<RQ, 256, 0, stream>>>(Ao, g_out, b_out, Aol);

  // 6) out = LN(attn) @ Wo^T + bo + latents
  gemm_wmma_kernel<true><<<dim3(DIMC / 128, RQ / 128), 256, 0, stream>>>(
      Aol, wob, nullptr, out, bo, lat);
}
